// RegionProposalNetwork_2723009265769
// MI455X (gfx1250) — matrix operand order repaired and checked by reading, not compiled, not run on hardware
//
#include <hip/hip_runtime.h>
#include <hip/hip_bf16.h>
#include <math.h>

#define HH 64
#define WW 64
#define NB 4
#define CIN 512
#define COUT 512
#define S_TOT (HH*WW*9)     /* 36864 */
#define P_TOT (HH*WW)       /* 4096  */
#define PRE_NMSK 2000
#define POST_NMSK 300
#define SORT_N 65536
#define WPR 63              /* ceil(2000/32) */
#define NMS_TH 0.7f
#define MIN_SZ 16.0f
#define KCH 144             /* 4608/32 K-chunks for the conv */

typedef __attribute__((ext_vector_type(16))) __bf16 v16bf;
typedef __attribute__((ext_vector_type(8)))  float  v8f;

union FragBF { uint4 u[2]; v16bf v; };

// ---- CDNA5 async global->LDS staging (ASYNCcnt) with sync fallback ----------
// Use inline asm for the load (bypasses clang's builtin pointer-AS typing);
// completion via s_wait_asynccnt builtin (probe-confirmed to exist & compile).
#if defined(__has_builtin)
# if __has_builtin(__builtin_amdgcn_s_wait_asynccnt)
#  define HAVE_ASYNC_LDS 1
# endif
#endif
#ifndef HAVE_ASYNC_LDS
# define HAVE_ASYNC_LDS 0
#endif

__device__ __forceinline__ void copy16_g2l(unsigned short* ldst, const unsigned short* gsrc) {
#if HAVE_ASYNC_LDS
  // VDST = per-lane LDS byte address (generic LDS addr[31:0] == LDS offset);
  // VADDR = 64-bit global address. Tracked by ASYNCcnt.
  const unsigned lds_off = (unsigned)(__SIZE_TYPE__)ldst;
  const unsigned long long gaddr = (unsigned long long)(__SIZE_TYPE__)gsrc;
  asm volatile("global_load_async_to_lds_b128 %0, %1, off"
               :
               : "v"(lds_off), "v"(gaddr)
               : "memory");
#else
  *(uint4*)ldst = *(const uint4*)gsrc;
#endif
}
__device__ __forceinline__ void async_fence_all() {
#if HAVE_ASYNC_LDS
  __builtin_amdgcn_s_wait_asynccnt(0);
#endif
}

__device__ __forceinline__ unsigned short f2bf(float f) {
  unsigned u = __float_as_uint(f);
  u += 0x7fffu + ((u >> 16) & 1u);          // round-to-nearest-even
  return (unsigned short)(u >> 16);
}

// ---------------- pack x: NCHW f32 -> padded (66x66) NHWC bf16 ----------------
__global__ void pack_x_kernel(const float* __restrict__ x, unsigned short* __restrict__ xp) {
  long long i = (long long)blockIdx.x * blockDim.x + threadIdx.x;
  const long long tot = (long long)NB * 66 * 66 * CIN;
  if (i >= tot) return;
  int ic = (int)(i & (CIN - 1));
  int t  = (int)(i >> 9);
  int xx = t % 66; t /= 66;
  int yy = t % 66; int n = t / 66;
  float v = 0.f;
  if (yy >= 1 && yy <= HH && xx >= 1 && xx <= WW)
    v = x[(((long long)n * CIN + ic) * HH + (yy - 1)) * WW + (xx - 1)];
  xp[i] = f2bf(v);
}

// ------- pack W1 into B-fragment order: [chunk 144][octile 32][lane 32][e 16] -
// B 32x16 bf16 K order used here (contiguous): lane half kh selects K base (0/16); element e -> K=kh*16+e
__global__ void pack_w_kernel(const float* __restrict__ W1, unsigned short* __restrict__ wpf) {
  int i = blockIdx.x * blockDim.x + threadIdx.x;
  if (i >= (KCH << 14)) return;
  const int e  = i & 15;
  const int l  = (i >> 4) & 31;
  const int t  = (i >> 9) & 31;
  const int c  = i >> 14;
  const int Kl = (l >> 4) * 16 + e;           // B-fragment K mapping
  const int k  = c * 32 + Kl;                 // 0..4607, k = (ky*3+kx)*512 + ic
  const int s  = k >> 9;
  const int ic = k & 511;
  const int ky = s / 3, kx = s - ky * 3;
  const int oc = t * 16 + (l & 15);
  wpf[i] = f2bf(W1[(((long long)oc * CIN + ic) * 3 + ky) * 3 + kx]);
}

// ------- pack heads (Wl rows 0..35, Ws rows 36..53, zero 54..63) in B-frag order
__global__ void pack_heads_kernel(const float* __restrict__ Wl, const float* __restrict__ Ws,
                                  unsigned short* __restrict__ whf) {
  int i = blockIdx.x * blockDim.x + threadIdx.x;
  if (i >= 16 * 4 * 32 * 16) return;
  const int e = i & 15;
  const int l = (i >> 4) & 31;
  const int t = (i >> 9) & 3;
  const int c = i >> 11;
  const int K = c * 32 + (l >> 4) * 16 + e;
  const int o = t * 16 + (l & 15);
  float v = 0.f;
  if (o < 36)      v = Wl[o * 512 + K];
  else if (o < 54) v = Ws[(o - 36) * 512 + K];
  whf[i] = f2bf(v);
}

// ---------------- implicit-GEMM 3x3 conv via WMMA bf16 -----------------------
// block: 256 thr (8 waves). block tile: 128 pixels x 128 ocs. Each wave: 32x64
// (2 A-frags x 4 B-frags -> 8 WMMAs/K-chunk, B reused across both M halves).
// Double-buffered LDS staged with async global->LDS; 1 barrier per K-chunk.
__global__ __launch_bounds__(256) void conv3x3_wmma_kernel(
    const unsigned short* __restrict__ xp,   // [NB][66][66][512]
    const unsigned short* __restrict__ wpf,  // fragment-ordered weights
    const float* __restrict__ b1,
    unsigned short* __restrict__ hbuf) {     // [NB*4096][512] bf16
  __shared__ unsigned short lsA[2][128][32]; // 16 KB  (pixel x ic)
  __shared__ unsigned short lsB[2][4096];    // 16 KB  (8 octiles, fragment order)
  const int tid  = threadIdx.x;
  const int lane = tid & 31;
  const int wave = tid >> 5;
  const int wm   = wave & 3;                 // M quarter (32 pixels)
  const int wn   = wave >> 2;                // N half (64 ocs)
  const int n    = blockIdx.z;
  const int pm0  = blockIdx.x * 128;
  const int gt0  = blockIdx.y * 8;           // octile base (8 per block)
  const int m    = lane & 15;
  const int kh   = lane >> 4;

  const int rA = tid >> 1;                   // 0..127 pixel row
  const int cA = (tid & 1) * 16;             // ic sub-offset
  const int pA = pm0 + rA;
  const int yA = pA >> 6, xA = pA & 63;

  v8f acc0[4] = {};
  v8f acc1[4] = {};

  auto stage = [&](int c, int buf) {
    const int kk  = c * 32;
    const int s   = kk >> 9;
    const int ic0 = kk & 511;
    const int ky  = s / 3, kx = s - ky * 3;
    const unsigned short* ga =
        &xp[((((long long)n * 66 + (yA + ky)) * 66 + (xA + kx)) << 9) + ic0 + cA];
    copy16_g2l(&lsA[buf][rA][cA],     ga);
    copy16_g2l(&lsA[buf][rA][cA + 8], ga + 8);
    const long long wb = ((long long)(c * 32 + gt0) << 9) + tid * 16;
    copy16_g2l(&lsB[buf][tid * 16],     &wpf[wb]);
    copy16_g2l(&lsB[buf][tid * 16 + 8], &wpf[wb + 8]);
  };

  stage(0, 0);
  for (int c = 0; c < KCH; ++c) {
    async_fence_all();                       // our async stage landed in LDS
    __syncthreads();                         // everyone's stage visible; prior reads done
    if (c + 1 < KCH) {
      stage(c + 1, (c + 1) & 1);
      if (c + 2 < KCH)                       // pull next weight tile toward L2
        __builtin_prefetch(&wpf[((long long)((c + 2) * 32 + gt0) << 9) + tid * 16], 0, 1);
    }
    const int buf = c & 1;
    FragBF a0, a1;                           // A 16x32 layout: K = kh*16 + e (contiguous)
    a0.u[0] = *(const uint4*)&lsA[buf][wm * 32 + m][kh * 16];
    a0.u[1] = *(const uint4*)&lsA[buf][wm * 32 + m][kh * 16 + 8];
    a1.u[0] = *(const uint4*)&lsA[buf][wm * 32 + 16 + m][kh * 16];
    a1.u[1] = *(const uint4*)&lsA[buf][wm * 32 + 16 + m][kh * 16 + 8];
    #pragma unroll
    for (int t = 0; t < 4; ++t) {
      FragBF bfr;
      const int off = ((wn * 4 + t) * 32 + lane) * 16;
      bfr.u[0] = *(const uint4*)&lsB[buf][off];
      bfr.u[1] = *(const uint4*)&lsB[buf][off + 8];
      acc0[t] = __builtin_amdgcn_wmma_f32_16x16x32_bf16(false, a0.v, false, bfr.v,
                                                        (short)0, acc0[t], false, false);
      acc1[t] = __builtin_amdgcn_wmma_f32_16x16x32_bf16(false, a1.v, false, bfr.v,
                                                        (short)0, acc1[t], false, false);
    }
  }
  #pragma unroll
  for (int t = 0; t < 4; ++t) {
    const int oc = (gt0 + wn * 4 + t) * 16 + m;
    const float bias = b1[oc];
    #pragma unroll
    for (int r = 0; r < 8; ++r) {
      const int p0 = pm0 + wm * 32 + r + 8 * kh;   // C layout: VGPR r -> M=r (+8 hi lanes)
      float v0 = acc0[t][r] + bias;
      v0 = v0 > 0.f ? v0 : 0.f;
      hbuf[(((long long)n * P_TOT + p0) << 9) + oc] = f2bf(v0);
      const int p1 = p0 + 16;
      float v1 = acc1[t][r] + bias;
      v1 = v1 > 0.f ? v1 : 0.f;
      hbuf[(((long long)n * P_TOT + p1) << 9) + oc] = f2bf(v1);
    }
  }
}

// ---------------- heads as WMMA GEMM: 16384 x 64 x 512 -----------------------
__global__ __launch_bounds__(256) void heads_wmma_kernel(
    const unsigned short* __restrict__ hbuf,  // [16384][512] bf16
    const unsigned short* __restrict__ whf,   // [16][4][32][16] bf16
    const float* __restrict__ bl, const float* __restrict__ bs,
    float* __restrict__ o_locs, float* __restrict__ o_scores) {
  extern __shared__ unsigned short lsW[];     // 32768 elems = 64 KB
  const int tid = threadIdx.x;
  #pragma unroll
  for (int j = 0; j < 8; ++j) {
    const int idx = j * 4096 + tid * 16;
    copy16_g2l(&lsW[idx],     &whf[idx]);
    copy16_g2l(&lsW[idx + 8], &whf[idx + 8]);
  }
  async_fence_all();
  __syncthreads();
  const int lane = tid & 31;
  const int wave = tid >> 5;
  const int m  = lane & 15;
  const int kh = lane >> 4;
  const int gt = blockIdx.x * 8 + wave;       // pixel tile 0..1023
  const unsigned short* arow = hbuf + ((long long)(gt * 16 + m) << 9);
  v8f acc[4] = {};
  for (int c = 0; c < 16; ++c) {
    FragBF a;
    const unsigned short* ap = arow + c * 32;
    a.u[0] = *(const uint4*)(ap + kh * 16);
    a.u[1] = *(const uint4*)(ap + kh * 16 + 8);
    #pragma unroll
    for (int t = 0; t < 4; ++t) {
      FragBF bfr;
      const int off = ((c * 4 + t) * 32 + lane) * 16;
      bfr.u[0] = *(const uint4*)&lsW[off];
      bfr.u[1] = *(const uint4*)&lsW[off + 8];
      acc[t] = __builtin_amdgcn_wmma_f32_16x16x32_bf16(false, a.v, false, bfr.v,
                                                       (short)0, acc[t], false, false);
    }
  }
  #pragma unroll
  for (int t = 0; t < 4; ++t) {
    const int o = t * 16 + m;                 // 0..63 (54..63 padding)
    #pragma unroll
    for (int r = 0; r < 8; ++r) {
      const int gp = gt * 16 + r + 8 * kh;
      const int n  = gp >> 12;
      const int p  = gp & 4095;
      float v = acc[t][r];
      if (o < 36) {
        v += bl[o];
        o_locs[(((long long)n * S_TOT + p * 9 + (o >> 2)) << 2) + (o & 3)] = v;
      } else if (o < 54) {
        const int o2 = o - 36;
        v += bs[o2];
        o_scores[((long long)n * S_TOT + p * 9 + (o2 >> 1)) * 2 + (o2 & 1)] = v;
      }
    }
  }
}

// ---------------- anchors ----------------------------------------------------
__global__ void anchors_kernel(float* __restrict__ o_anch) {
  int s = blockIdx.x * blockDim.x + threadIdx.x;
  if (s >= S_TOT) return;
  const int a = s % 9;
  const int p = s / 9;
  const int yc = p >> 6, xc = p & 63;
  const float ratios[3] = {0.5f, 1.0f, 2.0f};
  const float scales[3] = {8.0f, 16.0f, 32.0f};
  const int ri = a / 3, si = a - ri * 3;
  const float h = 16.0f * scales[si] * sqrtf(ratios[ri]);
  const float w = 16.0f * scales[si] * sqrtf(1.0f / ratios[ri]);
  const float py = 8.0f + (float)yc * 16.0f;
  const float px = 8.0f + (float)xc * 16.0f;
  float* o = o_anch + (long long)s * 4;
  o[0] = py - 0.5f * h; o[1] = px - 0.5f * w;
  o[2] = py + 0.5f * h; o[3] = px + 0.5f * w;
}

// ------- decode + clip + min-size + softmax fg + sort-key init ---------------
__global__ void decode_kernel(const float* __restrict__ o_locs,
                              const float* __restrict__ o_scores,
                              const float* __restrict__ o_anch,
                              const float* __restrict__ img_info,
                              float* __restrict__ boxes,
                              float* __restrict__ skey, int* __restrict__ sidx) {
  int i = blockIdx.x * blockDim.x + threadIdx.x;
  if (i >= NB * SORT_N) return;
  const int img = i >> 16;
  const int s   = i & (SORT_N - 1);
  const float NEGINF = -__builtin_huge_valf();
  if (s >= S_TOT) { skey[i] = NEGINF; sidx[i] = 0; return; }
  const float* an = o_anch + (long long)s * 4;
  const float ah = an[2] - an[0], aw = an[3] - an[1];
  const float acy = an[0] + 0.5f * ah, acx = an[1] + 0.5f * aw;
  const float* lc = o_locs + ((long long)img * S_TOT + s) * 4;
  const float cy = lc[0] * ah + acy;
  const float cx = lc[1] * aw + acx;
  const float bh = __expf(lc[2]) * ah;
  const float bw = __expf(lc[3]) * aw;
  const float H = img_info[0], W = img_info[1];
  float y1 = fminf(fmaxf(cy - 0.5f * bh, 0.f), H);
  float x1 = fminf(fmaxf(cx - 0.5f * bw, 0.f), W);
  float y2 = fminf(fmaxf(cy + 0.5f * bh, 0.f), H);
  float x2 = fminf(fmaxf(cx + 0.5f * bw, 0.f), W);
  float* bo = boxes + ((long long)img * S_TOT + s) * 4;
  bo[0] = y1; bo[1] = x1; bo[2] = y2; bo[3] = x2;
  const float* sco = o_scores + ((long long)img * S_TOT + s) * 2;
  const float fgv = 1.f / (1.f + __expf(sco[0] - sco[1]));   // softmax fg prob
  const bool valid = ((y2 - y1) >= MIN_SZ) && ((x2 - x1) >= MIN_SZ);
  skey[i] = valid ? fgv : NEGINF;
  sidx[i] = s;
}

// ---------------- global bitonic sort (descending), per image ----------------
__global__ void bitonic_kernel(float* __restrict__ key, int* __restrict__ idx, int k, int j) {
  int g = blockIdx.x * blockDim.x + threadIdx.x;
  const int img = g >> 16;
  const int t   = g & (SORT_N - 1);
  const int ixj = t ^ j;
  if (ixj <= t) return;
  const int base = img << 16;
  float ka = key[base + t], kb = key[base + ixj];
  const bool up = ((t & k) == 0);
  bool swp = up ? (ka < kb) : (ka > kb);
  if (swp) {
    key[base + t] = kb; key[base + ixj] = ka;
    int ia = idx[base + t]; idx[base + t] = idx[base + ixj]; idx[base + ixj] = ia;
  }
}

// ---------------- gather top-2000 sorted boxes -------------------------------
__global__ void gather_kernel(const float* __restrict__ key, const int* __restrict__ idx,
                              const float* __restrict__ boxes,
                              float* __restrict__ bsort, float* __restrict__ ssc) {
  int g = blockIdx.x * blockDim.x + threadIdx.x;
  if (g >= NB * PRE_NMSK) return;
  const int img = g / PRE_NMSK;
  const int i   = g - img * PRE_NMSK;
  const float scv = key[(img << 16) + i];
  const int s = idx[(img << 16) + i];
  ssc[g] = scv;
  float4 b4 = make_float4(0.f, 0.f, 0.f, 0.f);
  if (scv > -1e37f) {
    const float* bo = boxes + ((long long)img * S_TOT + s) * 4;
    b4 = make_float4(bo[0], bo[1], bo[2], bo[3]);
  }
  *(float4*)&bsort[(long long)g * 4] = b4;
}

// ---------------- NMS suppression bitmask ------------------------------------
__global__ void nms_mask_kernel(const float* __restrict__ bsort, unsigned* __restrict__ mask) {
  int g = blockIdx.x * blockDim.x + threadIdx.x;
  if (g >= NB * PRE_NMSK * WPR) return;
  int img = g / (PRE_NMSK * WPR);
  int rem = g - img * (PRE_NMSK * WPR);
  int i   = rem / WPR;
  int jw  = rem - i * WPR;
  const float* bi = bsort + ((long long)(img * PRE_NMSK + i)) * 4;
  const float iy1 = bi[0], ix1 = bi[1], iy2 = bi[2], ix2 = bi[3];
  const float ai = (iy2 - iy1) * (ix2 - ix1);
  unsigned bits = 0;
  for (int b = 0; b < 32; ++b) {
    int jj = jw * 32 + b;
    if (jj >= PRE_NMSK || jj <= i) continue;
    const float* bj = bsort + ((long long)(img * PRE_NMSK + jj)) * 4;
    float yy1 = fmaxf(iy1, bj[0]);
    float xx1 = fmaxf(ix1, bj[1]);
    float yy2 = fminf(iy2, bj[2]);
    float xx2 = fminf(ix2, bj[3]);
    float inter = fmaxf(yy2 - yy1, 0.f) * fmaxf(xx2 - xx1, 0.f);
    float aj = (bj[2] - bj[0]) * (bj[3] - bj[1]);
    float iou = inter / (ai + aj - inter + 1e-9f);
    if (iou > NMS_TH) bits |= (1u << b);
  }
  mask[g] = bits;
}

// ---------------- sequential scan + emit rois/indices ------------------------
__global__ void nms_scan_kernel(const unsigned* __restrict__ mask,
                                const float* __restrict__ ssc,
                                const float* __restrict__ bsort,
                                float* __restrict__ o_rois, float* __restrict__ o_roiidx) {
  const int img = threadIdx.x;
  if (img >= NB) return;
  unsigned remv[WPR];
  for (int w = 0; w < WPR; ++w) remv[w] = 0u;
  int cnt = 0;
  for (int i = 0; i < PRE_NMSK && cnt < POST_NMSK; ++i) {
    const int w = i >> 5, b = i & 31;
    const bool alive = (ssc[img * PRE_NMSK + i] > -1e37f) && !((remv[w] >> b) & 1u);
    if (!alive) continue;
    const float* bo = bsort + ((long long)(img * PRE_NMSK + i)) * 4;
    float* ro = o_rois + ((long long)(img * POST_NMSK + cnt)) * 4;
    ro[0] = bo[0]; ro[1] = bo[1]; ro[2] = bo[2]; ro[3] = bo[3];
    cnt++;
    const unsigned* mr = mask + (long long)(img * PRE_NMSK + i) * WPR;
    for (int w2 = 0; w2 < WPR; ++w2) remv[w2] |= mr[w2];
  }
  for (int r = cnt; r < POST_NMSK; ++r) {
    float* ro = o_rois + ((long long)(img * POST_NMSK + r)) * 4;
    ro[0] = 0.f; ro[1] = 0.f; ro[2] = 0.f; ro[3] = 0.f;
  }
  for (int r = 0; r < POST_NMSK; ++r) o_roiidx[img * POST_NMSK + r] = (float)img;
}

extern "C" void kernel_launch(void* const* d_in, const int* in_sizes, int n_in,
                              void* d_out, int out_size, void* d_ws, size_t ws_size,
                              hipStream_t stream) {
  (void)in_sizes; (void)n_in; (void)out_size; (void)ws_size;
  const float* x   = (const float*)d_in[0];
  const float* img = (const float*)d_in[1];   // device [H, W]
  const float* W1  = (const float*)d_in[3];
  const float* b1  = (const float*)d_in[4];
  const float* Ws  = (const float*)d_in[5];
  const float* bs  = (const float*)d_in[6];
  const float* Wl  = (const float*)d_in[7];
  const float* bl  = (const float*)d_in[8];

  float* out = (float*)d_out;
  float* o_locs   = out;                                       // NB*S*4
  float* o_scores = o_locs   + (long long)NB * S_TOT * 4;      // NB*S*2
  float* o_rois   = o_scores + (long long)NB * S_TOT * 2;      // NB*300*4
  float* o_roiidx = o_rois   + NB * POST_NMSK * 4;             // NB*300
  float* o_anch   = o_roiidx + NB * POST_NMSK;                 // S*4

  char* wsp = (char*)d_ws;
  auto alloc = [&](size_t bytes) {
    char* p = wsp; wsp += (bytes + 255) & ~(size_t)255; return p;
  };
  unsigned short* xp   = (unsigned short*)alloc((size_t)NB * 66 * 66 * CIN * 2);
  unsigned short* wpf  = (unsigned short*)alloc((size_t)(KCH << 14) * 2);
  unsigned short* whf  = (unsigned short*)alloc((size_t)16 * 4 * 32 * 16 * 2);
  unsigned short* hbuf = (unsigned short*)alloc((size_t)NB * P_TOT * COUT * 2);
  float*    boxes = (float*)   alloc((size_t)NB * S_TOT * 4 * 4);
  float*    skey  = (float*)   alloc((size_t)NB * SORT_N * 4);
  int*      sidx  = (int*)     alloc((size_t)NB * SORT_N * 4);
  float*    bsort = (float*)   alloc((size_t)NB * PRE_NMSK * 4 * 4);
  float*    ssc   = (float*)   alloc((size_t)NB * PRE_NMSK * 4);
  unsigned* mask  = (unsigned*)alloc((size_t)NB * PRE_NMSK * WPR * 4);

  { long long tot = (long long)NB * 66 * 66 * CIN;
    pack_x_kernel<<<(int)((tot + 255) / 256), 256, 0, stream>>>(x, xp); }
  pack_w_kernel<<<((KCH << 14) + 255) / 256, 256, 0, stream>>>(W1, wpf);
  pack_heads_kernel<<<(32768 + 255) / 256, 256, 0, stream>>>(Wl, Ws, whf);
  { dim3 g(P_TOT / 128, COUT / 128, NB);
    conv3x3_wmma_kernel<<<g, 256, 0, stream>>>(xp, wpf, b1, hbuf); }
  heads_wmma_kernel<<<128, 256, 65536, stream>>>(hbuf, whf, bl, bs, o_locs, o_scores);
  anchors_kernel<<<(S_TOT + 255) / 256, 256, 0, stream>>>(o_anch);
  decode_kernel<<<(NB * SORT_N) / 256, 256, 0, stream>>>(o_locs, o_scores, o_anch, img,
                                                         boxes, skey, sidx);
  for (int k = 2; k <= SORT_N; k <<= 1)
    for (int j = k >> 1; j > 0; j >>= 1)
      bitonic_kernel<<<(NB * SORT_N) / 256, 256, 0, stream>>>(skey, sidx, k, j);
  gather_kernel<<<(NB * PRE_NMSK + 255) / 256, 256, 0, stream>>>(skey, sidx, boxes, bsort, ssc);
  nms_mask_kernel<<<(NB * PRE_NMSK * WPR + 255) / 256, 256, 0, stream>>>(bsort, mask);
  nms_scan_kernel<<<1, 32, 0, stream>>>(mask, ssc, bsort, o_rois, o_roiidx);
}
